// Decoder_26834955665940
// MI455X (gfx1250) — matrix operand order repaired and checked by reading, not compiled, not run on hardware
//
#include <hip/hip_runtime.h>

#define DEV __device__ __forceinline__

typedef __attribute__((ext_vector_type(16))) _Float16 v16h;
typedef __attribute__((ext_vector_type(8)))  _Float16 v8h_t;
typedef __attribute__((ext_vector_type(8)))  float    v8f;

// Problem constants
#define BB   32
#define TT   512
#define LL   400
#define EE   256
#define DD   80
#define HH   512
#define XX   768   // H + E
#define KM   5

DEV v16h cat8(v8h_t lo, v8h_t hi) {
  return __builtin_shufflevector(lo, hi, 0,1,2,3,4,5,6,7,8,9,10,11,12,13,14,15);
}

// A fragment from a lane-resolved pointer (row*ld + hi-lane half-offset already
// applied): halves 0..7 at p, halves 8..15 at p+16 (ISA 16-bit A layout).
DEV v16h ldA(const _Float16* p) {
  return cat8(*(const v8h_t*)p, *(const v8h_t*)(p + 16));
}
// B fragment from a lane-resolved pointer: halves 0..7 at p, halves 8..15 at p+16.
DEV v16h ldB(const _Float16* p) {
  return cat8(*(const v8h_t*)p, *(const v8h_t*)(p + 16));
}

DEV v8f wmma_f16(v16h a, v16h b, v8f c) {
  return __builtin_amdgcn_wmma_f32_16x16x32_f16(false, a, false, b, (short)0, c,
                                                false, false);
}

DEV float sigm(float x) { return 1.0f / (1.0f + __expf(-x)); }
DEV float softplusf(float x) { return (x > 20.0f) ? x : log1pf(__expf(x)); }

// ---------------- prep kernels ----------------
__global__ void k_cvt(const float* __restrict__ s, _Float16* __restrict__ d, int n) {
  int i = blockIdx.x * blockDim.x + threadIdx.x;
  if (i < n) d[i] = (_Float16)s[i];
}
__global__ void k_zero(unsigned int* __restrict__ p, int nWords) {
  int i = blockIdx.x * blockDim.x + threadIdx.x;
  if (i < nWords) p[i] = 0u;
}

// ---------------- GRU: h' = GRUCell([h,ctx], h) ----------------
// grid = H/16 blocks, 64 threads (2 waves; wave = M tile). Block owns h-cols [16*bx,+16).
__global__ void __launch_bounds__(64) k_gru(
    const _Float16* __restrict__ h16p, const _Float16* __restrict__ ctx16p,
    const _Float16* __restrict__ wih16, const _Float16* __restrict__ whh16,
    const float* __restrict__ bih, const float* __restrict__ bhh,
    float* __restrict__ hf, _Float16* __restrict__ h16c) {
  const int nBase = blockIdx.x * 16;
  const int mBase = (threadIdx.x >> 5) * 16;
  const int lane  = threadIdx.x & 31;
  const int aHi   = (lane & 16) >> 1;  // +8 halves on hi lanes (A layout)
  const int bHi   = (lane & 16) >> 1;  // +8 halves on hi lanes (B layout)
  const int aRow  = mBase + (lane & 15);
  const int bRow  = nBase + (lane & 15);

  // lane-resolved base pointers; K-loop offsets fold into load immediates
  const _Float16* aH  = h16p  + (size_t)aRow * HH + aHi;
  const _Float16* aC  = ctx16p + (size_t)aRow * EE + aHi;
  const _Float16* bRi = wih16 + (size_t)bRow * XX + bHi;
  const _Float16* bZi = wih16 + (size_t)(bRow + HH) * XX + bHi;
  const _Float16* bNi = wih16 + (size_t)(bRow + 2 * HH) * XX + bHi;
  const _Float16* bRh = whh16 + (size_t)bRow * HH + bHi;
  const _Float16* bZh = whh16 + (size_t)(bRow + HH) * HH + bHi;
  const _Float16* bNh = whh16 + (size_t)(bRow + 2 * HH) * HH + bHi;

  v8f accR = {}, accZ = {}, accN = {}, accHN = {};
  // x @ Wih^T, x = [h(0:512) | ctx(512:768)]  -- split, no runtime selects
  for (int k = 0; k < HH; k += 32) {
    v16h a = ldA(aH + k);
    accR = wmma_f16(a, ldB(bRi + k), accR);
    accZ = wmma_f16(a, ldB(bZi + k), accZ);
    accN = wmma_f16(a, ldB(bNi + k), accN);
  }
  for (int k = 0; k < EE; k += 32) {
    v16h a = ldA(aC + k);
    accR = wmma_f16(a, ldB(bRi + HH + k), accR);
    accZ = wmma_f16(a, ldB(bZi + HH + k), accZ);
    accN = wmma_f16(a, ldB(bNi + HH + k), accN);
  }
  // h @ Whh^T
  for (int k = 0; k < HH; k += 32) {
    v16h a = ldA(aH + k);
    accR  = wmma_f16(a, ldB(bRh + k), accR);
    accZ  = wmma_f16(a, ldB(bZh + k), accZ);
    accHN = wmma_f16(a, ldB(bNh + k), accHN);
  }

  const int col = nBase + (lane & 15);
  const int rowOff = aHi;  // 0 or 8 (C/D layout row offset for hi lanes)
  const float bR  = bih[col] + bhh[col];
  const float bZ  = bih[col + HH] + bhh[col + HH];
  const float bIN = bih[col + 2 * HH];
  const float bHN = bhh[col + 2 * HH];
#pragma unroll
  for (int i = 0; i < 8; ++i) {
    const int m = mBase + rowOff + i;
    float r  = sigm(accR[i] + bR);
    float z  = sigm(accZ[i] + bZ);
    float nn = tanhf(accN[i] + bIN + r * (accHN[i] + bHN));
    float hold = hf[m * HH + col];
    float hn = (1.0f - z) * nn + z * hold;
    hf[m * HH + col] = hn;
    h16c[m * HH + col] = (_Float16)hn;
  }
}

// ---------------- MoL attention + context (one block per batch) ----------------
__global__ void __launch_bounds__(256) k_att(
    const _Float16* __restrict__ h16c, const _Float16* __restrict__ aw1_16,
    const float* __restrict__ ab1, const float* __restrict__ aw2,
    const float* __restrict__ ab2, const _Float16* __restrict__ enc16,
    float* __restrict__ muG, _Float16* __restrict__ ctx16c,
    float* __restrict__ alphas, int l) {
  __shared__ float hs[HH];
  __shared__ float as[128];
  __shared__ float ps[3 * KM];
  __shared__ float wk[KM], muk[KM], sck[KM];
  __shared__ float al[TT];
  const int b = blockIdx.x;
  const int tid = threadIdx.x;
  for (int k = tid; k < HH; k += 256) hs[k] = (float)h16c[b * HH + k];
  __syncthreads();
  if (tid < 128) {
    const _Float16* wr = aw1_16 + (size_t)tid * HH;
    float acc = ab1[tid];
    for (int k = 0; k < HH; k += 8) {
      v8h_t w8 = *(const v8h_t*)(wr + k);
#pragma unroll
      for (int j = 0; j < 8; ++j) acc += hs[k + j] * (float)w8[j];
    }
    as[tid] = tanhf(acc);
  }
  __syncthreads();
  if (tid < 3 * KM) {
    const float* wr = aw2 + tid * 128;
    float acc = ab2[tid];
    for (int k = 0; k < 128; ++k) acc += as[k] * wr[k];
    ps[tid] = acc;
  }
  __syncthreads();
  if (tid == 0) {
    float mx = -1e30f;
    for (int k = 0; k < KM; ++k) {
      mx = fmaxf(mx, ps[k]);
      float mu = muG[b * KM + k] + softplusf(ps[KM + k]);
      muG[b * KM + k] = mu;
      muk[k] = mu;
      sck[k] = softplusf(ps[2 * KM + k]) + 1e-4f;
    }
    float s = 0.0f;
    for (int k = 0; k < KM; ++k) { wk[k] = __expf(ps[k] - mx); s += wk[k]; }
    for (int k = 0; k < KM; ++k) wk[k] /= s;
  }
  __syncthreads();
  for (int t = tid; t < TT; t += 256) {
    float acc = 0.0f;
#pragma unroll
    for (int k = 0; k < KM; ++k) {
      float inv = 1.0f / sck[k];
      float zc = ((float)t - muk[k]) * inv;
      float hf2 = 0.5f * inv;
      acc += wk[k] * (sigm(zc + hf2) - sigm(zc - hf2));
    }
    al[t] = acc;
    alphas[((size_t)b * LL + l) * TT + t] = acc;
  }
  __syncthreads();
  {
    const int e = tid;  // 256 threads == E
    const _Float16* ez = enc16 + (size_t)b * TT * EE + e;
    float acc = 0.0f;
    for (int t = 0; t < TT; ++t) acc += al[t] * (float)ez[(size_t)t * EE];
    ctx16c[b * EE + e] = (_Float16)acc;
  }
}

// ---------------- LSTM: (lh,lc) = LSTMCell([ctx,h], lh, lc) ----------------
__global__ void __launch_bounds__(64) k_lstm(
    const _Float16* __restrict__ ctx16c, const _Float16* __restrict__ h16c,
    const _Float16* __restrict__ lh16p,
    const _Float16* __restrict__ wih16, const _Float16* __restrict__ whh16,
    const float* __restrict__ bih, const float* __restrict__ bhh,
    float* __restrict__ lcf, _Float16* __restrict__ lh16c) {
  const int nBase = blockIdx.x * 16;
  const int mBase = (threadIdx.x >> 5) * 16;
  const int lane  = threadIdx.x & 31;
  const int aHi   = (lane & 16) >> 1;
  const int bHi   = (lane & 16) >> 1;
  const int aRow  = mBase + (lane & 15);
  const int bRow  = nBase + (lane & 15);

  const _Float16* aC  = ctx16c + (size_t)aRow * EE + aHi;
  const _Float16* aH  = h16c  + (size_t)aRow * HH + aHi;
  const _Float16* aL  = lh16p + (size_t)aRow * HH + aHi;
  const _Float16* bIi = wih16 + (size_t)bRow * XX + bHi;
  const _Float16* bFi = wih16 + (size_t)(bRow + HH) * XX + bHi;
  const _Float16* bGi = wih16 + (size_t)(bRow + 2 * HH) * XX + bHi;
  const _Float16* bOi = wih16 + (size_t)(bRow + 3 * HH) * XX + bHi;
  const _Float16* bIh = whh16 + (size_t)bRow * HH + bHi;
  const _Float16* bFh = whh16 + (size_t)(bRow + HH) * HH + bHi;
  const _Float16* bGh = whh16 + (size_t)(bRow + 2 * HH) * HH + bHi;
  const _Float16* bOh = whh16 + (size_t)(bRow + 3 * HH) * HH + bHi;

  v8f aI = {}, aF = {}, aG = {}, aO = {};
  // li @ Wih^T, li = [ctx(0:256) | h(256:768)]
  for (int k = 0; k < EE; k += 32) {
    v16h a = ldA(aC + k);
    aI = wmma_f16(a, ldB(bIi + k), aI);
    aF = wmma_f16(a, ldB(bFi + k), aF);
    aG = wmma_f16(a, ldB(bGi + k), aG);
    aO = wmma_f16(a, ldB(bOi + k), aO);
  }
  for (int k = 0; k < HH; k += 32) {
    v16h a = ldA(aH + k);
    aI = wmma_f16(a, ldB(bIi + EE + k), aI);
    aF = wmma_f16(a, ldB(bFi + EE + k), aF);
    aG = wmma_f16(a, ldB(bGi + EE + k), aG);
    aO = wmma_f16(a, ldB(bOi + EE + k), aO);
  }
  // lh @ Whh^T
  for (int k = 0; k < HH; k += 32) {
    v16h a = ldA(aL + k);
    aI = wmma_f16(a, ldB(bIh + k), aI);
    aF = wmma_f16(a, ldB(bFh + k), aF);
    aG = wmma_f16(a, ldB(bGh + k), aG);
    aO = wmma_f16(a, ldB(bOh + k), aO);
  }

  const int col = nBase + (lane & 15);
  const int rowOff = aHi;
  const float bI = bih[col] + bhh[col];
  const float bF = bih[col + HH] + bhh[col + HH];
  const float bG = bih[col + 2 * HH] + bhh[col + 2 * HH];
  const float bO = bih[col + 3 * HH] + bhh[col + 3 * HH];
#pragma unroll
  for (int i = 0; i < 8; ++i) {
    const int m = mBase + rowOff + i;
    float cold = lcf[m * HH + col];
    float cn = sigm(aF[i] + bF) * cold + sigm(aI[i] + bI) * tanhf(aG[i] + bG);
    float hn = sigm(aO[i] + bO) * tanhf(cn);
    lcf[m * HH + col] = cn;
    lh16c[m * HH + col] = (_Float16)hn;
  }
}

// ---------------- out layer 1: u = relu(lh @ W1^T + b1), (32 x 80) ----------------
__global__ void __launch_bounds__(64) k_out1(
    const _Float16* __restrict__ lh16c, const _Float16* __restrict__ w1_16,
    const float* __restrict__ b1, float* __restrict__ u) {
  const int nBase = blockIdx.x * 16;  // 5 blocks cover D=80
  const int mBase = (threadIdx.x >> 5) * 16;
  const int lane  = threadIdx.x & 31;
  const int aHi   = (lane & 16) >> 1;
  const int bHi   = (lane & 16) >> 1;
  const _Float16* aL = lh16c + (size_t)(mBase + (lane & 15)) * HH + aHi;
  const _Float16* bW = w1_16 + (size_t)(nBase + (lane & 15)) * HH + bHi;
  v8f acc = {};
  for (int k = 0; k < HH; k += 32)
    acc = wmma_f16(ldA(aL + k), ldB(bW + k), acc);
  const int col = nBase + (lane & 15);
  const int rowOff = aHi;
  const float bb = b1[col];
#pragma unroll
  for (int i = 0; i < 8; ++i) {
    const int m = mBase + rowOff + i;
    u[m * DD + col] = fmaxf(acc[i] + bb, 0.0f);
  }
}

// ---------------- out layer 2 + stop logit (one block per batch) ----------------
__global__ void __launch_bounds__(128) k_out2(
    const float* __restrict__ u, const float* __restrict__ w2,
    const float* __restrict__ b2, const float* __restrict__ sw,
    const float* __restrict__ sb, float* __restrict__ outs0,
    float* __restrict__ outs1, float* __restrict__ logits, int l) {
  __shared__ float ys[DD];
  const int b = blockIdx.x;
  const int tid = threadIdx.x;
  if (tid < DD) {
    const float* ub = u + b * DD;
    const float* wr = w2 + tid * DD;
    float acc = b2[tid];
    for (int k = 0; k < DD; ++k) acc += ub[k] * wr[k];
    ys[tid] = acc;
    size_t o = ((size_t)b * LL + l) * DD + tid;
    outs0[o] = acc;
    outs1[o] = acc;
  }
  __syncthreads();
  if (tid == 0) {
    float acc = sb[0];
    for (int d = 0; d < DD; ++d) acc += ys[d] * sw[d];
    logits[(size_t)b * LL + l] = acc;
  }
}

extern "C" void kernel_launch(void* const* d_in, const int* in_sizes, int n_in,
                              void* d_out, int out_size, void* d_ws, size_t ws_size,
                              hipStream_t stream) {
  (void)in_sizes; (void)n_in; (void)out_size; (void)ws_size;
  const float* enc_z    = (const float*)d_in[0];
  const float* gru_wih  = (const float*)d_in[2];
  const float* gru_whh  = (const float*)d_in[3];
  const float* gru_bih  = (const float*)d_in[4];
  const float* gru_bhh  = (const float*)d_in[5];
  const float* att_w1   = (const float*)d_in[6];
  const float* att_b1   = (const float*)d_in[7];
  const float* att_w2   = (const float*)d_in[8];
  const float* att_b2   = (const float*)d_in[9];
  const float* lstm_wih = (const float*)d_in[10];
  const float* lstm_whh = (const float*)d_in[11];
  const float* lstm_bih = (const float*)d_in[12];
  const float* lstm_bhh = (const float*)d_in[13];
  const float* out_w1   = (const float*)d_in[14];
  const float* out_b1   = (const float*)d_in[15];
  const float* out_w2   = (const float*)d_in[16];
  const float* out_b2   = (const float*)d_in[17];
  const float* stop_w   = (const float*)d_in[18];
  const float* stop_b   = (const float*)d_in[19];

  float* outp   = (float*)d_out;
  float* outs0  = outp;
  float* outs1  = outp + (size_t)BB * LL * DD;
  float* logits = outp + (size_t)2 * BB * LL * DD;
  float* alphas = logits + (size_t)BB * LL;

  char* ws = (char*)d_ws;
  size_t off = 0;
  auto alloc = [&](size_t bytes) -> void* {
    void* p = ws + off;
    off = (off + bytes + 255) & ~(size_t)255;
    return p;
  };

  // f16 weight copies + enc_z f16
  _Float16* wgih16 = (_Float16*)alloc((size_t)3 * HH * XX * 2);
  _Float16* wghh16 = (_Float16*)alloc((size_t)3 * HH * HH * 2);
  _Float16* wlih16 = (_Float16*)alloc((size_t)4 * HH * XX * 2);
  _Float16* wlhh16 = (_Float16*)alloc((size_t)4 * HH * HH * 2);
  _Float16* wa1_16 = (_Float16*)alloc((size_t)128 * HH * 2);
  _Float16* wo1_16 = (_Float16*)alloc((size_t)DD * HH * 2);
  _Float16* enc16  = (_Float16*)alloc((size_t)BB * TT * EE * 2);

  // recurrent state (zeroed every launch) — contiguous for one zero kernel
  size_t zero_begin = off;
  _Float16* h16[2]  = { (_Float16*)alloc((size_t)BB * HH * 2),
                        (_Float16*)alloc((size_t)BB * HH * 2) };
  _Float16* ctx16[2]= { (_Float16*)alloc((size_t)BB * EE * 2),
                        (_Float16*)alloc((size_t)BB * EE * 2) };
  _Float16* lh16[2] = { (_Float16*)alloc((size_t)BB * HH * 2),
                        (_Float16*)alloc((size_t)BB * HH * 2) };
  float* hf  = (float*)alloc((size_t)BB * HH * 4);
  float* lcf = (float*)alloc((size_t)BB * HH * 4);
  float* mu  = (float*)alloc((size_t)BB * KM * 4);
  size_t zero_end = off;
  float* u   = (float*)alloc((size_t)BB * DD * 4);

  auto cvt = [&](const float* s, _Float16* d, int n) {
    k_cvt<<<(n + 255) / 256, 256, 0, stream>>>(s, d, n);
  };
  cvt(gru_wih,  wgih16, 3 * HH * XX);
  cvt(gru_whh,  wghh16, 3 * HH * HH);
  cvt(lstm_wih, wlih16, 4 * HH * XX);
  cvt(lstm_whh, wlhh16, 4 * HH * HH);
  cvt(att_w1,   wa1_16, 128 * HH);
  cvt(out_w1,   wo1_16, DD * HH);
  cvt(enc_z,    enc16,  BB * TT * EE);

  int zWords = (int)((zero_end - zero_begin) / 4);
  k_zero<<<(zWords + 255) / 256, 256, 0, stream>>>(
      (unsigned int*)(ws + zero_begin), zWords);

  for (int l = 0; l < LL; ++l) {
    const int pr = l & 1;
    const int cu = pr ^ 1;
    k_gru<<<HH / 16, 64, 0, stream>>>(h16[pr], ctx16[pr], wgih16, wghh16,
                                      gru_bih, gru_bhh, hf, h16[cu]);
    k_att<<<BB, 256, 0, stream>>>(h16[cu], wa1_16, att_b1, att_w2, att_b2,
                                  enc16, mu, ctx16[cu], alphas, l);
    k_lstm<<<HH / 16, 64, 0, stream>>>(ctx16[cu], h16[cu], lh16[pr], wlih16,
                                       wlhh16, lstm_bih, lstm_bhh, lcf, lh16[cu]);
    k_out1<<<DD / 16, 64, 0, stream>>>(lh16[cu], wo1_16, out_b1, u);
    k_out2<<<BB, 128, 0, stream>>>(u, out_w2, out_b2, stop_w, stop_b,
                                   outs0, outs1, logits, l);
  }
}
